// MSCENAFormer_81724637708881
// MI455X (gfx1250) — compile-verified, matrix operand order checked by reading, not run on hardware
//
#include <hip/hip_runtime.h>
#include <hip/hip_bf16.h>

// ---------------------------------------------------------------------------
// MSCENAFormer pipeline for gfx1250 (MI455X, wave32, WMMA + TDM)
// B=4, C=96, H=W=64, HID=384, HEADS=6, dh=16, P=B*H*W=16384
// ---------------------------------------------------------------------------

typedef __attribute__((ext_vector_type(16))) _Float16 v16h;
typedef __attribute__((ext_vector_type(8)))  _Float16 v8h;
typedef __attribute__((ext_vector_type(8)))  float    v8f;
typedef unsigned int uint32x4 __attribute__((ext_vector_type(4)));
typedef int          int32x4  __attribute__((ext_vector_type(4)));
typedef int          int32x8  __attribute__((ext_vector_type(8)));

#define CC   96
#define HH   64
#define WW   64
#define BB   4
#define HID  384
#define NHEAD 6
#define DH   16
#define PIX  (BB*HH*WW)        // 16384
#define PC   (PIX*CC)          // 1572864
#define EPSV 1e-5f

__device__ __forceinline__ float gelu_f(float x) {
    return 0.5f * x * (1.f + erff(x * 0.70710678118654752f));
}

// ---------------------------------------------------------------------------
// TDM: issue a 2D tensor tile load (f16 elements) global -> LDS.
//   gptr      : tile start in global memory (f16*)
//   lds_off   : byte offset of destination within workgroup LDS
//   td0/td1   : tensor dims (elements) for OOB
//   tile0/1   : tile dims (elements); tile0 = contiguous (K) direction
//   stride0   : tensor dim0 stride (elements)
// Descriptor bitfields per CDNA5 ISA 8.3/8.4 (group0 128b, group1 256b).
// Toolchain note: this clang exposes the 6-arg builtin
//   (uint32x4 g0, int32x8 g1, int32x4, int32x4, int32x8, i32 cpol)
// ---------------------------------------------------------------------------
#if __has_builtin(__builtin_amdgcn_tensor_load_to_lds) && \
    __has_builtin(__builtin_amdgcn_s_wait_tensorcnt)
#define USE_TDM 1
#else
#define USE_TDM 0
#endif

#if USE_TDM
__device__ __forceinline__ void tdm_load_2d(const _Float16* gptr, unsigned lds_off,
                                            unsigned td0, unsigned td1,
                                            unsigned tile0, unsigned tile1,
                                            unsigned stride0)
{
    unsigned long long ga = (unsigned long long)gptr;
    uint32x4 g0;
    g0[0] = 1u;                                            // count=1, user mode
    g0[1] = lds_off;                                       // lds_addr [63:32]
    g0[2] = (unsigned)(ga & 0xFFFFFFFFu);                  // global_addr lo
    g0[3] = (unsigned)((ga >> 32) & 0x01FFFFFFu)           // global_addr hi (57b)
          | (2u << 30);                                    // type=2 (image)
    int32x8 g1;
    g1[0] = (int)(1u << 16);                               // data_size=1 (2 bytes)
    g1[1] = (int)((td0 & 0xFFFFu) << 16);                  // tensor_dim0 lo16 @ [63:48]
    g1[2] = (int)((td0 >> 16) | ((td1 & 0xFFFFu) << 16));  // dim0 hi16 | dim1 lo16
    g1[3] = (int)((td1 >> 16) | (tile0 << 16));            // dim1 hi16 | tile_dim0
    g1[4] = (int)(tile1 & 0xFFFFu);                        // tile_dim1 (tile_dim2=0)
    g1[5] = (int)stride0;                                  // dim0_stride lo32
    g1[6] = 0;                                             // dim0_stride hi16 | dim1_stride lo16
    g1[7] = 0;
    int32x4 z4 = {0, 0, 0, 0};
    int32x8 z8 = {0, 0, 0, 0, 0, 0, 0, 0};
    __builtin_amdgcn_tensor_load_to_lds(g0, g1, z4, z4, z8, 0);
}
#endif

// ---------------------------------------------------------------------------
// WMMA GEMM (f16 operands, f32 accum): C[M,N] = act(A[M,K] @ Bt^T + bias)
//   A  : [M,K] f16 row-major
//   Bt : [N,K] f16 row-major (weights pre-transposed/converted once)
//   ACT: 0=none 1=relu 2=gelu
// Block = 256 threads (8 wave32s). Block tile 64(M) x 32(N), K-step 32.
// Tiles staged into LDS by the Tensor Data Mover (wave 0 issues, TENSORcnt
// wait, then workgroup barrier publishes). Fragments read as ds_load_b128.
// Requires M%64==0, N%32==0, K%32==0 (true for all uses here).
// ---------------------------------------------------------------------------
template<int ACT>
__global__ __launch_bounds__(256)
void gemm_wmma_tdm(const _Float16* __restrict__ A, const _Float16* __restrict__ Bt,
                   const float* __restrict__ bias, float* __restrict__ Cm,
                   int M, int N, int K)
{
    // Contiguous tiles: As = 64 rows x 32 halfs, Bs = 32 rows x 32 halfs.
    __shared__ _Float16 As[64 * 32];
    __shared__ _Float16 Bs[32 * 32];

    const int tid  = threadIdx.x;
    const int wave = tid >> 5;
    const int lane = tid & 31;
    const int wm   = wave & 3;    // M subtile 0..3
    const int wn   = wave >> 2;   // N subtile 0..1
    const int m0   = blockIdx.y * 64;
    const int n0   = blockIdx.x * 32;

    v8f acc = {0.f,0.f,0.f,0.f,0.f,0.f,0.f,0.f};

    const int half = lane >> 4;
    const int l15  = lane & 15;
    const int ar   = wm * 16 + l15;   // A row within tile
    const int br   = wn * 16 + l15;   // Bt row (= output column) within tile

    for (int k0 = 0; k0 < K; k0 += 32) {
#if USE_TDM
        if (wave == 0) {
            // A tile: 64 rows x 32 halfs starting at (m0, k0)
            tdm_load_2d(A  + (size_t)m0 * K + k0, 0u,
                        (unsigned)K, (unsigned)M, 32u, 64u, (unsigned)K);
            // B tile: 32 rows x 32 halfs starting at (n0, k0)
            tdm_load_2d(Bt + (size_t)n0 * K + k0, 64u * 32u * 2u,
                        (unsigned)K, (unsigned)N, 32u, 32u, (unsigned)K);
            __builtin_amdgcn_s_wait_tensorcnt(0);
        }
        __syncthreads();
#else
        for (int i = tid; i < 64 * 32; i += 256) {
            int r = i >> 5, c = i & 31;
            As[i] = A[(size_t)(m0 + r) * K + (k0 + c)];
        }
        for (int i = tid; i < 32 * 32; i += 256) {
            int r = i >> 5, c = i & 31;
            Bs[i] = Bt[(size_t)(n0 + r) * K + (k0 + c)];
        }
        __syncthreads();
#endif
        // Fragment load per ISA 7.12.2: lanes 0-15 hold K 0..7 / 16..23,
        // lanes 16-31 hold K 8..15 / 24..31. Two contiguous b128 chunks each.
        v8h a_lo = *(const v8h*)&As[ar * 32 + half * 8];
        v8h a_hi = *(const v8h*)&As[ar * 32 + 16 + half * 8];
        v8h b_lo = *(const v8h*)&Bs[br * 32 + half * 8];
        v8h b_hi = *(const v8h*)&Bs[br * 32 + 16 + half * 8];
        v16h af = __builtin_shufflevector(a_lo, a_hi,
                    0,1,2,3,4,5,6,7,8,9,10,11,12,13,14,15);
        v16h bf = __builtin_shufflevector(b_lo, b_hi,
                    0,1,2,3,4,5,6,7,8,9,10,11,12,13,14,15);
        acc = __builtin_amdgcn_wmma_f32_16x16x32_f16(
                  false, af, false, bf, (short)0, acc, false, false);
        __syncthreads();
    }

    // Epilogue: C/D layout -> lane l, vgpr v: M = v + 8*(l>>4), N = l&15
    const int col = n0 + wn * 16 + l15;
    const float bv = bias ? bias[col] : 0.f;
    #pragma unroll
    for (int v = 0; v < 8; ++v) {
        int row = m0 + wm * 16 + v + 8 * half;
        float val = acc[v] + bv;
        if (ACT == 1) val = fmaxf(val, 0.f);
        else if (ACT == 2) val = gelu_f(val);
        Cm[(size_t)row * N + col] = val;
    }
}

// ---------------------------------------------------------------------------
// Weight prep: f32 -> f16 in [N,K] layout.
//   TRANS=0: w is [K,N] row-major (x@w style)  -> out[n*K+k] = w[k*N+n]
//   TRANS=1: w is [N,K] row-major (w[o,c] conv)-> out[n*K+k] = w[n*K+k]
// ---------------------------------------------------------------------------
template<int TRANS>
__global__ void weight_prep(const float* __restrict__ w, _Float16* __restrict__ out,
                            int N, int K)
{
    int i = blockIdx.x * blockDim.x + threadIdx.x;
    if (i >= N * K) return;
    int n = i / K, k = i % K;
    float v = TRANS ? w[(size_t)n * K + k] : w[(size_t)k * N + n];
    out[i] = (_Float16)v;
}

__global__ void convert_f16(const float* __restrict__ in, _Float16* __restrict__ out, int n)
{
    int i = blockIdx.x * blockDim.x + threadIdx.x;
    if (i < n) out[i] = (_Float16)in[i];
}

// ---------------------------------------------------------------------------
// rLN: per-batch mean/std over C*H*W
// ---------------------------------------------------------------------------
__global__ void rln_stats(const float* __restrict__ X, float* __restrict__ stats)
{
    __shared__ float ss[256], sq[256];
    const int nper = CC * HH * WW;
    const float* xb = X + (size_t)blockIdx.x * nper;
    float s = 0.f, q = 0.f;
    for (int i = threadIdx.x; i < nper; i += 256) {
        float v = xb[i]; s += v; q += v * v;
    }
    ss[threadIdx.x] = s; sq[threadIdx.x] = q;
    __syncthreads();
    for (int st = 128; st > 0; st >>= 1) {
        if (threadIdx.x < st) { ss[threadIdx.x] += ss[threadIdx.x + st];
                                sq[threadIdx.x] += sq[threadIdx.x + st]; }
        __syncthreads();
    }
    if (threadIdx.x == 0) {
        float mean = ss[0] / nper;
        float var  = sq[0] / nper - mean * mean;
        stats[blockIdx.x * 2]     = mean;
        stats[blockIdx.x * 2 + 1] = sqrtf(var + EPSV);
    }
}

// Y_nhwc[p,c] = (X_nchw - mean)/std * ln_w[c] + ln_b[c]
__global__ void rln_apply(const float* __restrict__ X, const float* __restrict__ stats,
                          const float* __restrict__ lnw, const float* __restrict__ lnb,
                          float* __restrict__ Y)
{
    int idx = blockIdx.x * blockDim.x + threadIdx.x;   // over B*C*H*W (NCHW)
    int w = idx % WW;
    int h = (idx / WW) % HH;
    int c = (idx / (WW * HH)) % CC;
    int b = idx / (CC * HH * WW);
    float mean = stats[b * 2], std = stats[b * 2 + 1];
    float val = (X[idx] - mean) / std * lnw[c] + lnb[c];
    int p = (b * HH + h) * WW + w;
    Y[(size_t)p * CC + c] = val;
}

__global__ void rln_scales(const float* __restrict__ stats,
                           const float* __restrict__ m1w, const float* __restrict__ m1b,
                           const float* __restrict__ m2w, const float* __restrict__ m2b,
                           float* __restrict__ rs, float* __restrict__ rb)
{
    int i = blockIdx.x * blockDim.x + threadIdx.x;     // over B*C
    if (i >= BB * CC) return;
    int b = i / CC, c = i % CC;
    float mean = stats[b * 2], std = stats[b * 2 + 1];
    rs[i] = m1w[c] * std + m1b[c];
    rb[i] = m2w[c] * mean + m2b[c];
}

// out_nchw = sc_nchw + (useScale ? t_nhwc*rs + rb : t_nhwc)
__global__ void residual_kernel(const float* __restrict__ sc, const float* __restrict__ t,
                                const float* __restrict__ rs, const float* __restrict__ rb,
                                float* __restrict__ out, int useScale)
{
    int idx = blockIdx.x * blockDim.x + threadIdx.x;   // NCHW
    int w = idx % WW;
    int h = (idx / WW) % HH;
    int c = (idx / (WW * HH)) % CC;
    int b = idx / (CC * HH * WW);
    int p = (b * HH + h) * WW + w;
    float v = t[(size_t)p * CC + c];
    float r = useScale ? (v * rs[b * CC + c] + rb[b * CC + c]) : v;
    out[idx] = sc[idx] + r;
}

// ---------------------------------------------------------------------------
// Neighborhood attention core: qkv[P,288] layout [3][head][dh]; out[P,96].
// One thread per (pixel, head); fully unrolled KS*KS taps kept in VGPRs.
// ---------------------------------------------------------------------------
template<int KS, int DIL>
__global__ __launch_bounds__(256)
void na_kernel(const float* __restrict__ qkv, const float* __restrict__ rpb,
               float* __restrict__ out)
{
    int tid  = blockIdx.x * blockDim.x + threadIdx.x;  // B*H*W*NHEAD
    int head = tid % NHEAD;
    int p    = tid / NHEAD;
    int y    = p % WW;
    int x    = (p / WW) % HH;
    int pb   = (p / (HH * WW)) * (HH * WW);

    const float* qp = qkv + (size_t)p * (3 * CC) + head * DH;
    float q[DH];
    #pragma unroll
    for (int d = 0; d < DH; ++d) q[d] = qp[d] * 0.25f;   // dh^-0.5 = 16^-0.5

    int gx = x % DIL, px = x / DIL;
    int Lgx = (HH - gx + DIL - 1) / DIL;
    int sx = px - KS / 2; sx = sx < 0 ? 0 : sx; if (sx > Lgx - KS) sx = Lgx - KS;
    int gy = y % DIL, py = y / DIL;
    int Lgy = (WW - gy + DIL - 1) / DIL;
    int sy = py - KS / 2; sy = sy < 0 ? 0 : sy; if (sy > Lgy - KS) sy = Lgy - KS;

    const int RK = 2 * KS - 1;
    const float* rp = rpb + head * RK * RK;

    float s[KS * KS];
    float mx = -1e30f;
    #pragma unroll
    for (int i = 0; i < KS; ++i) {
        int nx = gx + DIL * (sx + i);
        int rh = sx + i - px + (KS - 1);
        #pragma unroll
        for (int j = 0; j < KS; ++j) {
            int ny = gy + DIL * (sy + j);
            int rw = sy + j - py + (KS - 1);
            const float* kp = qkv + (size_t)(pb + nx * WW + ny) * (3 * CC) + CC + head * DH;
            float dot = 0.f;
            #pragma unroll
            for (int d = 0; d < DH; ++d) dot += q[d] * kp[d];
            dot += rp[rh * RK + rw];
            s[i * KS + j] = dot;
            mx = fmaxf(mx, dot);
        }
    }
    float sum = 0.f;
    #pragma unroll
    for (int t = 0; t < KS * KS; ++t) { s[t] = __expf(s[t] - mx); sum += s[t]; }
    float inv = 1.f / sum;

    float acc[DH];
    #pragma unroll
    for (int d = 0; d < DH; ++d) acc[d] = 0.f;
    #pragma unroll
    for (int i = 0; i < KS; ++i) {
        int nx = gx + DIL * (sx + i);
        #pragma unroll
        for (int j = 0; j < KS; ++j) {
            int ny = gy + DIL * (sy + j);
            const float* vp = qkv + (size_t)(pb + nx * WW + ny) * (3 * CC) + 2 * CC + head * DH;
            float wgt = s[i * KS + j] * inv;
            #pragma unroll
            for (int d = 0; d < DH; ++d) acc[d] += wgt * vp[d];
        }
    }
    float* op = out + (size_t)p * CC + head * DH;
    #pragma unroll
    for (int d = 0; d < DH; ++d) op[d] = acc[d];
}

// ---------------------------------------------------------------------------
// Block 4 helpers
// ---------------------------------------------------------------------------
__global__ void bn_kernel(const float* __restrict__ X,
                          const float* __restrict__ g, const float* __restrict__ bb,
                          const float* __restrict__ m, const float* __restrict__ v,
                          float* __restrict__ Y)
{
    int idx = blockIdx.x * blockDim.x + threadIdx.x;   // NCHW
    int w = idx % WW;
    int h = (idx / WW) % HH;
    int c = (idx / (WW * HH)) % CC;
    int b = idx / (CC * HH * WW);
    float inv = rsqrtf(v[c] + EPSV);
    float val = (X[idx] - m[c]) * inv * g[c] + bb[c];
    int p = (b * HH + h) * WW + w;
    Y[(size_t)p * CC + c] = val;
}

// pixel attention: pa = sigmoid(w2 @ gelu(w1 @ y + b1) + b2); Y *= pa (in place)
__global__ void pa_kernel(float* __restrict__ Y,
                          const float* __restrict__ w1, const float* __restrict__ b1,
                          const float* __restrict__ w2, const float* __restrict__ b2)
{
    const int CH = CC / 8;  // 12
    int p = blockIdx.x * blockDim.x + threadIdx.x;
    float* yr = Y + (size_t)p * CC;
    float accum = b2[0];
    for (int o = 0; o < CH; ++o) {
        float a = b1[o];
        for (int c = 0; c < CC; ++c) a += yr[c] * w1[o * CC + c];
        accum += gelu_f(a) * w2[o];
    }
    float pa = 1.f / (1.f + __expf(-accum));
    for (int c = 0; c < CC; ++c) yr[c] *= pa;
}

__global__ void gap_kernel(const float* __restrict__ Y, float* __restrict__ gap)
{
    __shared__ float sh[256];
    const int HW = HH * WW;
    int bc = blockIdx.x;                 // over B*C
    int b = bc / CC, c = bc % CC;
    const float* base = Y + (size_t)b * HW * CC + c;
    float s = 0.f;
    for (int i = threadIdx.x; i < HW; i += 256) s += base[(size_t)i * CC];
    sh[threadIdx.x] = s;
    __syncthreads();
    for (int st = 128; st > 0; st >>= 1) {
        if (threadIdx.x < st) sh[threadIdx.x] += sh[threadIdx.x + st];
        __syncthreads();
    }
    if (threadIdx.x == 0) gap[bc] = sh[0] / HW;
}

// channel attention on gap[b,:]: ca = sigmoid(w2 @ gelu(w1 @ gap + b1) + b2)
__global__ void ca_kernel(const float* __restrict__ gap,
                          const float* __restrict__ w1, const float* __restrict__ b1,
                          const float* __restrict__ w2, const float* __restrict__ b2,
                          float* __restrict__ ca)
{
    __shared__ float g[CC], h1[CC];
    int b = blockIdx.x, o = threadIdx.x;
    if (o < CC) g[o] = gap[b * CC + o];
    __syncthreads();
    if (o < CC) {
        float a = b1[o];
        for (int c = 0; c < CC; ++c) a += g[c] * w1[o * CC + c];
        h1[o] = gelu_f(a);
    }
    __syncthreads();
    if (o < CC) {
        float a = b2[o];
        for (int c = 0; c < CC; ++c) a += h1[c] * w2[o * CC + c];
        ca[b * CC + o] = 1.f / (1.f + __expf(-a));
    }
}

__global__ void scale_ca_kernel(float* __restrict__ Y, const float* __restrict__ ca)
{
    int idx = blockIdx.x * blockDim.x + threadIdx.x;   // over P*C (NHWC)
    int p = idx / CC, c = idx % CC;
    int b = p / (HH * WW);
    Y[idx] *= ca[b * CC + c];
}

// ---------------------------------------------------------------------------
extern "C" void kernel_launch(void* const* d_in, const int* in_sizes, int n_in,
                              void* d_out, int out_size, void* d_ws, size_t ws_size,
                              hipStream_t stream)
{
    (void)in_sizes; (void)n_in; (void)out_size; (void)ws_size;

    const float* x      = (const float*)d_in[0];
    const float* ln_w   = (const float*)d_in[1];
    const float* ln_b   = (const float*)d_in[2];
    const float* m1_w   = (const float*)d_in[3];
    const float* m1_b   = (const float*)d_in[4];
    const float* m2_w   = (const float*)d_in[5];
    const float* m2_b   = (const float*)d_in[6];
    const float* qkv1_w = (const float*)d_in[7];
    const float* qkv1_b = (const float*)d_in[8];
    const float* proj1_w= (const float*)d_in[9];
    const float* proj1_b= (const float*)d_in[10];
    const float* rpb1   = (const float*)d_in[11];
    const float* qkv2_w = (const float*)d_in[12];
    const float* qkv2_b = (const float*)d_in[13];
    const float* proj2_w= (const float*)d_in[14];
    const float* proj2_b= (const float*)d_in[15];
    const float* rpb2   = (const float*)d_in[16];
    const float* mlp_w1 = (const float*)d_in[17];
    const float* mlp_b1 = (const float*)d_in[18];
    const float* mlp_w2 = (const float*)d_in[19];
    const float* mlp_b2 = (const float*)d_in[20];
    const float* bn_g   = (const float*)d_in[21];
    const float* bn_b   = (const float*)d_in[22];
    const float* bn_m   = (const float*)d_in[23];
    const float* bn_v   = (const float*)d_in[24];
    const float* pa_w1  = (const float*)d_in[25];
    const float* pa_b1  = (const float*)d_in[26];
    const float* pa_w2  = (const float*)d_in[27];
    const float* pa_b2  = (const float*)d_in[28];
    const float* ca_w1  = (const float*)d_in[29];
    const float* ca_b1  = (const float*)d_in[30];
    const float* ca_w2  = (const float*)d_in[31];
    const float* ca_b2  = (const float*)d_in[32];
    const float* m2a_w1 = (const float*)d_in[33];
    const float* m2a_b1 = (const float*)d_in[34];
    const float* m2a_w2 = (const float*)d_in[35];
    const float* m2a_b2 = (const float*)d_in[36];

    float* outp = (float*)d_out;

    // workspace layout
    float* X     = (float*)d_ws;        // PC : running state (NCHW)
    float* SC    = X + PC;              // PC : skip copy (NCHW)
    float* Y     = SC + PC;             // PC : normalized / activation (NHWC)
    float* T1    = Y + PC;              // P*HID : qkv out / mlp hidden (NHWC)
    float* T2    = T1 + (size_t)PIX*HID;// PC : per-stage result (NHWC)
    float* stats = T2 + PC;             // 2*B
    float* rs    = stats + 2 * BB;      // B*C
    float* rb    = rs + BB * CC;        // B*C
    float* gap   = rb + BB * CC;        // B*C
    float* ca    = gap + BB * CC;       // B*C
    _Float16* Ah     = (_Float16*)(ca + BB * CC);   // P*HID halfs (GEMM A in f16)
    _Float16* Wqkv1  = Ah + (size_t)PIX * HID;      // 288*96
    _Float16* Wproj1 = Wqkv1 + 288 * CC;            // 96*96
    _Float16* Wqkv2  = Wproj1 + CC * CC;
    _Float16* Wproj2 = Wqkv2 + 288 * CC;
    _Float16* Wmlp1  = Wproj2 + CC * CC;            // 384*96
    _Float16* Wmlp2  = Wmlp1 + HID * CC;            // 96*384
    _Float16* Wm2a1  = Wmlp2 + CC * HID;            // 96*96
    _Float16* Wm2a2  = Wm2a1 + CC * CC;             // 96*96

    const int EW = 256;
    const int NB = PC / EW;             // 6144 blocks for P*C elementwise
    dim3 gQKV(288 / 32, PIX / 64);
    dim3 gC  (CC  / 32, PIX / 64);
    dim3 gHID(HID / 32, PIX / 64);
    const int NBH = (PIX * HID) / EW;   // convert grid for P*HID

    // ---- one-time weight conversion to f16 [N,K] ----
    weight_prep<0><<<(288*CC+255)/256, 256, 0, stream>>>(qkv1_w,  Wqkv1,  288, CC);
    weight_prep<0><<<(CC*CC+255)/256,  256, 0, stream>>>(proj1_w, Wproj1, CC,  CC);
    weight_prep<0><<<(288*CC+255)/256, 256, 0, stream>>>(qkv2_w,  Wqkv2,  288, CC);
    weight_prep<0><<<(CC*CC+255)/256,  256, 0, stream>>>(proj2_w, Wproj2, CC,  CC);
    weight_prep<1><<<(HID*CC+255)/256, 256, 0, stream>>>(mlp_w1,  Wmlp1,  HID, CC);
    weight_prep<1><<<(CC*HID+255)/256, 256, 0, stream>>>(mlp_w2,  Wmlp2,  CC,  HID);
    weight_prep<1><<<(CC*CC+255)/256,  256, 0, stream>>>(m2a_w1,  Wm2a1,  CC,  CC);
    weight_prep<1><<<(CC*CC+255)/256,  256, 0, stream>>>(m2a_w2,  Wm2a2,  CC,  CC);

    // x -> X
    (void)hipMemcpyAsync(X, x, (size_t)PC * sizeof(float), hipMemcpyDeviceToDevice, stream);

    // ---------------- Block 1: NA k=7 d=1 ----------------
    rln_stats<<<BB, 256, 0, stream>>>(X, stats);
    rln_apply<<<NB, EW, 0, stream>>>(X, stats, ln_w, ln_b, Y);
    rln_scales<<<(BB * CC + 255) / 256, 256, 0, stream>>>(stats, m1_w, m1_b, m2_w, m2_b, rs, rb);
    convert_f16<<<NB, EW, 0, stream>>>(Y, Ah, PC);
    gemm_wmma_tdm<0><<<gQKV, 256, 0, stream>>>(Ah, Wqkv1, qkv1_b, T1, PIX, 3 * CC, CC);
    na_kernel<7, 1><<<PIX * NHEAD / 256, 256, 0, stream>>>(T1, rpb1, Y);
    convert_f16<<<NB, EW, 0, stream>>>(Y, Ah, PC);
    gemm_wmma_tdm<0><<<gC, 256, 0, stream>>>(Ah, Wproj1, proj1_b, T2, PIX, CC, CC);
    residual_kernel<<<NB, EW, 0, stream>>>(X, T2, rs, rb, X, 1);

    // SC = x after block 1 (skip for block 3)
    (void)hipMemcpyAsync(SC, X, (size_t)PC * sizeof(float), hipMemcpyDeviceToDevice, stream);

    // ---------------- Block 2: NA k=5 d=8 ----------------
    rln_stats<<<BB, 256, 0, stream>>>(X, stats);
    rln_apply<<<NB, EW, 0, stream>>>(X, stats, ln_w, ln_b, Y);
    rln_scales<<<(BB * CC + 255) / 256, 256, 0, stream>>>(stats, m1_w, m1_b, m2_w, m2_b, rs, rb);
    convert_f16<<<NB, EW, 0, stream>>>(Y, Ah, PC);
    gemm_wmma_tdm<0><<<gQKV, 256, 0, stream>>>(Ah, Wqkv2, qkv2_b, T1, PIX, 3 * CC, CC);
    na_kernel<5, 8><<<PIX * NHEAD / 256, 256, 0, stream>>>(T1, rpb2, Y);
    convert_f16<<<NB, EW, 0, stream>>>(Y, Ah, PC);
    gemm_wmma_tdm<0><<<gC, 256, 0, stream>>>(Ah, Wproj2, proj2_b, T2, PIX, CC, CC);
    residual_kernel<<<NB, EW, 0, stream>>>(X, T2, rs, rb, X, 1);

    // ---------------- Block 3: MLP (skip from SC) ----------------
    rln_stats<<<BB, 256, 0, stream>>>(X, stats);
    rln_apply<<<NB, EW, 0, stream>>>(X, stats, ln_w, ln_b, Y);
    rln_scales<<<(BB * CC + 255) / 256, 256, 0, stream>>>(stats, m1_w, m1_b, m2_w, m2_b, rs, rb);
    convert_f16<<<NB, EW, 0, stream>>>(Y, Ah, PC);
    gemm_wmma_tdm<1><<<gHID, 256, 0, stream>>>(Ah, Wmlp1, mlp_b1, T1, PIX, HID, CC);   // relu
    convert_f16<<<NBH, EW, 0, stream>>>(T1, Ah, PIX * HID);
    gemm_wmma_tdm<0><<<gC, 256, 0, stream>>>(Ah, Wmlp2, mlp_b2, T2, PIX, CC, HID);
    residual_kernel<<<NB, EW, 0, stream>>>(SC, T2, rs, rb, X, 1);

    // ---------------- Block 4: BN + pixel/channel attention + MLP ----------------
    bn_kernel<<<NB, EW, 0, stream>>>(X, bn_g, bn_b, bn_m, bn_v, Y);
    pa_kernel<<<PIX / 256, 256, 0, stream>>>(Y, pa_w1, pa_b1, pa_w2, pa_b2);
    gap_kernel<<<BB * CC, 256, 0, stream>>>(Y, gap);
    ca_kernel<<<BB, CC, 0, stream>>>(gap, ca_w1, ca_b1, ca_w2, ca_b2, ca);
    scale_ca_kernel<<<NB, EW, 0, stream>>>(Y, ca);
    convert_f16<<<NB, EW, 0, stream>>>(Y, Ah, PC);
    gemm_wmma_tdm<2><<<gC, 256, 0, stream>>>(Ah, Wm2a1, m2a_b1, T1, PIX, CC, CC);      // gelu
    convert_f16<<<NB, EW, 0, stream>>>(T1, Ah, PC);
    gemm_wmma_tdm<0><<<gC, 256, 0, stream>>>(Ah, Wm2a2, m2a_b2, T2, PIX, CC, CC);
    residual_kernel<<<NB, EW, 0, stream>>>(X, T2, nullptr, nullptr, outp, 0);
}
